// BackendQueryPooling_27771258536728
// MI455X (gfx1250) — matrix operand order repaired and checked by reading, not compiled, not run on hardware
//
#include <hip/hip_runtime.h>
#include <hip/hip_bf16.h>
#include <stddef.h>

// ---------------------------------------------------------------------------
// BackendQueryPooling for MI455X (gfx1250, wave32, WMMA + async-to-LDS).
//
// Shapes: B=16, L=8192, D=256, H=8, hd=32, NB=8.
//   - Heavy work: K-proj (x @ Wk^T, 17 GFLOP) via v_wmma_f32_16x16x32_bf16.
//   - V-proj deferred: ctx = (sum_l p_l x_l) @ Wv^T + bv, so we accumulate
//     u = sum p*x with online softmax per (backend, head); Wv applied once in
//     the finalize kernel.
//   - Reference multiplies output by has_tokens, so the full-mask fallback is
//     unobservable -> dropped.
//   - x tiles staged with double-buffered GLOBAL_LOAD_ASYNC_TO_LDS_B128
//     (ASYNCcnt), overlapping HBM latency with WMMA compute.
// ---------------------------------------------------------------------------

#define DM 256
#define LX 8192
#define BX 16
#define HX 8
#define NBK 8
#define CHUNK 512
#define NCHUNK 16          // LX / CHUNK
#define TILES 32           // CHUNK / 16
#define WROW 264           // padded bf16 row stride (528B, 16B aligned)
#define PART_STRIDE 264    // floats per (chunk,head,slot) partial: m,s,u[256]
#define PART_OFF 264       // ws floats reserved for qh[256] + sbias[8]

typedef __bf16 v16bf  __attribute__((ext_vector_type(16)));
typedef __bf16 bf16x4 __attribute__((ext_vector_type(4)));
typedef float  v8f    __attribute__((ext_vector_type(8)));
typedef unsigned int u32x4 __attribute__((ext_vector_type(4)));

union FragBf { v16bf v; u32x4 q[2]; };

// ---- CDNA5 async global->LDS copy (per-lane 16B), ASYNCcnt tracked --------
#if __has_builtin(__builtin_amdgcn_global_load_async_to_lds_b128) && \
    __has_builtin(__builtin_amdgcn_s_wait_asynccnt)
#define HAVE_ASYNC_BUILTIN 1
#endif

// Parameter types per hipcc diagnostic: 'int __attribute__((vector_size(16))) __device__ *'
typedef int v4i_t __attribute__((vector_size(16)));
typedef __attribute__((address_space(1))) v4i_t* g_v4i_p;   // global
typedef __attribute__((address_space(3))) v4i_t* l_v4i_p;   // LDS

__device__ __forceinline__ void async_copy_b128(void* lds_dst, const void* gsrc)
{
#if defined(HAVE_ASYNC_BUILTIN)
    __builtin_amdgcn_global_load_async_to_lds_b128(
        (g_v4i_p)(unsigned long long)(size_t)gsrc,
        (l_v4i_p)(unsigned)(size_t)lds_dst,
        0, 0);
#else
    unsigned loff = (unsigned)(size_t)lds_dst;   // LDS byte offset for VDST
    asm volatile("global_load_async_to_lds_b128 %0, %1, off"
                 :: "v"(loff), "v"(gsrc) : "memory");
#endif
}

__device__ __forceinline__ void wait_async_zero()
{
#if defined(HAVE_ASYNC_BUILTIN)
    __builtin_amdgcn_s_wait_asynccnt(0);
#else
    asm volatile("s_wait_asynccnt 0x0" ::: "memory");
#endif
}

// ---------------------------------------------------------------------------
// Kernel 0: q projection (q = query @ Wq^T + bq) and per-head bias term
// sbias[h] = sum_d qh[h*32+d] * bk[h*32+d]  (added to every score pre-scale).
// ---------------------------------------------------------------------------
__global__ void __launch_bounds__(256)
bqp_prep(const float* __restrict__ query, const float* __restrict__ ipw,
         const float* __restrict__ ipb, float* __restrict__ ws)
{
    __shared__ float qv[DM];
    __shared__ float qh[DM];
    int tid = threadIdx.x;
    qv[tid] = query[tid];
    __syncthreads();
    float acc = ipb[tid];
    const float* wr = ipw + (size_t)tid * DM;   // Wq row tid
    #pragma unroll 4
    for (int d = 0; d < DM; ++d) acc += qv[d] * wr[d];
    qh[tid] = acc;
    ws[tid] = acc;
    __syncthreads();
    if (tid < HX) {
        float sb = 0.f;
        for (int j = 0; j < 32; ++j)
            sb += qh[tid * 32 + j] * ipb[DM + tid * 32 + j];   // bk slice
        ws[DM + tid] = sb;
    }
}

// ---------------------------------------------------------------------------
// Kernel 1: streaming K-proj (WMMA bf16) + online-softmax accumulation of
// u = sum p * x per (backend, head).  grid = BX * NCHUNK, block = 256 (8 waves,
// wave w owns head w).
// ---------------------------------------------------------------------------
__global__ void __launch_bounds__(256)
bqp_main(const float* __restrict__ x, const float* __restrict__ ipw,
         const unsigned char* __restrict__ maskp, const int* __restrict__ bidp,
         float* __restrict__ ws)
{
    __shared__ __bf16 wks[256][WROW];      // Wk (bf16)           132.0 KB
    __shared__ __bf16 xs[16][WROW];        // x tile (bf16)         8.25 KB
    __shared__ float  xstg[2][4096];       // async f32 staging    32 KB
    __shared__ float  kws[HX][16][33];     // per-wave k tile      16.5 KB
    __shared__ float  q_l[DM];             // projected q           1 KB
    __shared__ float  sb_l[HX];

    const int tid  = threadIdx.x;
    const int lane = tid & 31;
    const int w    = tid >> 5;             // wave id == head id
    const int b    = blockIdx.x >> 4;      // batch
    const int c    = blockIdx.x & 15;      // L-chunk

    const int l0 = c * CHUNK;
    const float* xb = x + (size_t)b * LX * DM;

    // ---- kick off async staging of tile 0 (overlaps with Wk load below) ----
    {
        const float* src = xb + (size_t)l0 * DM;
        #pragma unroll
        for (int rnd = 0; rnd < 4; ++rnd) {
            int fi = rnd * 256 + tid;              // float4 index (0..1023)
            async_copy_b128(&xstg[0][fi * 4], src + fi * 4);
        }
    }

    // ---- cache Wk (rows 256..511 of in_proj_weight) in LDS as bf16 ----
    {
        const float4* wkf = (const float4*)(ipw + 256 * DM);
        #pragma unroll 4
        for (int j = 0; j < 64; ++j) {
            int fi = j * 256 + tid;                // float4 index, 16384 total
            float4 v = wkf[fi];
            int e0 = fi * 4;
            int r = e0 >> 8, d = e0 & 255;
            bf16x4 pk = { (__bf16)v.x, (__bf16)v.y, (__bf16)v.z, (__bf16)v.w };
            *(bf16x4*)&wks[r][d] = pk;
        }
    }
    if (tid < DM) q_l[tid] = ws[tid];
    if (tid < HX) sb_l[tid] = ws[DM + tid];

    // ---- accumulators (registers, statically indexed) ----
    float um[NBK][8];
    float mx[NBK], sm[NBK];
    #pragma unroll
    for (int n = 0; n < NBK; ++n) {
        mx[n] = -1e30f; sm[n] = 0.f;
        #pragma unroll
        for (int i = 0; i < 8; ++i) um[n][i] = 0.f;
    }

    const int mrow = lane & 15, kb = lane >> 4;   // A-fragment mapping
    const int ncol = lane & 15, kh = lane >> 4;   // B-fragment mapping
    const int rbase = (lane >> 4) * 8;            // C-fragment row base

    for (int tile = 0; tile < TILES; ++tile) {
        const int cur = tile & 1;
        // current tile's async copies (issued last iteration / prologue) done
        wait_async_zero();
        // issue next tile into the other staging buffer (hidden by compute)
        if (tile + 1 < TILES) {
            const float* src = xb + (size_t)(l0 + (tile + 1) * 16) * DM;
            #pragma unroll
            for (int rnd = 0; rnd < 4; ++rnd) {
                int fi = rnd * 256 + tid;
                async_copy_b128(&xstg[cur ^ 1][fi * 4], src + fi * 4);
            }
        }
        __syncthreads();   // previous tile's xs fully consumed by all waves
        // ---- convert staged f32 tile -> bf16 xs (each thread: own bytes) ----
        #pragma unroll
        for (int rnd = 0; rnd < 4; ++rnd) {
            int fi = rnd * 256 + tid;
            float4 v = *(const float4*)&xstg[cur][fi * 4];
            int e0 = fi * 4;
            int r = e0 >> 8, d = e0 & 255;
            bf16x4 pk = { (__bf16)v.x, (__bf16)v.y, (__bf16)v.z, (__bf16)v.w };
            *(bf16x4*)&xs[r][d] = pk;
        }
        __syncthreads();

        const int lt = l0 + tile * 16;
        // ---- token metadata (lanes 0..15 hold token lane) ----
        int info = 0;
        if (lane < 16) {
            size_t gi = (size_t)b * LX + (lt + lane);
            unsigned char mk = maskp[gi];
            int bd = bidp[gi];
            info = mk ? (256 | bd) : 0;
        }

        // ---- WMMA K-projection: wave w computes output cols [32w, 32w+32) ----
        v8f c0 = {}; v8f c1 = {};
        #pragma unroll
        for (int ks = 0; ks < 8; ++ks) {
            FragBf a, b0, b1;
            // A (16x32 bf16): lane(kb*16+m): V0-3 = K[kb*8 .. +7], V4-7 = K[16+kb*8 .. +7]
            a.q[0] = *(const u32x4*)&xs[mrow][ks * 32 + kb * 8];
            a.q[1] = *(const u32x4*)&xs[mrow][ks * 32 + 16 + kb * 8];
            // B (32x16 bf16): lane(kh*16+n): V0-3 = K[kh*8 .. +7], V4-7 = K[16+kh*8 .. +7]
            b0.q[0] = *(const u32x4*)&wks[(2 * w) * 16 + ncol][ks * 32 + kh * 8];
            b0.q[1] = *(const u32x4*)&wks[(2 * w) * 16 + ncol][ks * 32 + 16 + kh * 8];
            b1.q[0] = *(const u32x4*)&wks[(2 * w + 1) * 16 + ncol][ks * 32 + kh * 8];
            b1.q[1] = *(const u32x4*)&wks[(2 * w + 1) * 16 + ncol][ks * 32 + 16 + kh * 8];
            c0 = __builtin_amdgcn_wmma_f32_16x16x32_bf16(false, a.v, false, b0.v,
                                                         (short)0, c0, false, false);
            c1 = __builtin_amdgcn_wmma_f32_16x16x32_bf16(false, a.v, false, b1.v,
                                                         (short)0, c1, false, false);
        }
        // C layout: lane holds column (lane&15), VGPR r -> row r + 8*(lane>>4)
        #pragma unroll
        for (int r = 0; r < 8; ++r) {
            kws[w][rbase + r][ncol]      = c0[r];
            kws[w][rbase + r][16 + ncol] = c1[r];
        }

        // ---- scores for this head (lanes 0..15 = tokens) ----
        float scl = 0.f;
        if (lane < 16) {
            float acc = sb_l[w];
            #pragma unroll
            for (int d2 = 0; d2 < 32; ++d2)
                acc += kws[w][lane][d2] * q_l[w * 32 + d2];
            scl = acc * 0.17677669529663689f;   // 1/sqrt(32)
        }

        // ---- online softmax accumulation of u = sum p*x ----
        #pragma unroll
        for (int t = 0; t < 16; ++t) {
            int inf = __shfl(info, t, 32);      // wave-uniform
            if (inf) {
                float sct = __shfl(scl, t, 32);
                int bd = inf & 255;
                float xv[8];
                #pragma unroll
                for (int i = 0; i < 8; ++i)
                    xv[i] = (float)xs[t][i * 32 + lane];
                #pragma unroll
                for (int n = 0; n < NBK; ++n) {
                    if (n == bd) {              // uniform branch, static indices
                        float mo = mx[n];
                        float mn = fmaxf(mo, sct);
                        float corr = __expf(mo - mn);
                        float p    = __expf(sct - mn);
                        mx[n] = mn;
                        sm[n] = sm[n] * corr + p;
                        #pragma unroll
                        for (int i = 0; i < 8; ++i)
                            um[n][i] = um[n][i] * corr + p * xv[i];
                    }
                }
            }
        }
    }

    // ---- write partials: ws[PART_OFF + ((block*8 + h)*8 + n)*PART_STRIDE] ----
    float* P = ws + PART_OFF + ((size_t)blockIdx.x * HX + w) * NBK * PART_STRIDE;
    #pragma unroll
    for (int n = 0; n < NBK; ++n) {
        float* Pn = P + n * PART_STRIDE;
        if (lane == 0) { Pn[0] = mx[n]; Pn[1] = sm[n]; }
        #pragma unroll
        for (int i = 0; i < 8; ++i)
            Pn[2 + i * 32 + lane] = um[n][i];
    }
}

// ---------------------------------------------------------------------------
// Kernel 2: merge chunk partials (log-sum-exp), ctx = (u/s)@Wv^T + bv,
// out-proj, LayerNorm, *has_tokens, transposed store to (B, NB, D).
// grid = NB*BX blocks of 256.
// ---------------------------------------------------------------------------
__global__ void __launch_bounds__(256)
bqp_final(const float* __restrict__ ipw, const float* __restrict__ ipb,
          const float* __restrict__ wo, const float* __restrict__ bo,
          const float* __restrict__ gamma, const float* __restrict__ beta,
          const float* __restrict__ ws, float* __restrict__ out)
{
    __shared__ float uu[HX][257];
    __shared__ float ctxs[DM];
    __shared__ float red[16];
    __shared__ float hasS, mu_s, rs_s;

    const int tid = threadIdx.x;
    const int n = blockIdx.x / BX;   // backend
    const int b = blockIdx.x % BX;   // batch
    const float* Pb = ws + PART_OFF;

    for (int h = 0; h < HX; ++h) {
        float M = -1e30f;
        for (int c2 = 0; c2 < NCHUNK; ++c2) {
            const float* Pn = Pb + (((size_t)(b * NCHUNK + c2) * HX + h) * NBK + n) * PART_STRIDE;
            M = fmaxf(M, Pn[0]);
        }
        float S = 0.f, U = 0.f;
        for (int c2 = 0; c2 < NCHUNK; ++c2) {
            const float* Pn = Pb + (((size_t)(b * NCHUNK + c2) * HX + h) * NBK + n) * PART_STRIDE;
            float e = __expf(Pn[0] - M);
            S += Pn[1] * e;
            U += Pn[2 + tid] * e;
        }
        uu[h][tid] = (S > 0.f) ? (U / S) : 0.f;
        if (h == 0 && tid == 0) hasS = S;
    }
    __syncthreads();

    const size_t obase = ((size_t)b * NBK + n) * DM;
    if (hasS <= 0.f) { out[obase + tid] = 0.f; return; }   // has_tokens == 0

    // ctx[j] = bv[j] + sum_d uu[h][d] * Wv[j][d]   (Wv row = ipw row 512+j)
    const int h = tid >> 5;
    float cj = ipb[512 + tid];
    const float* wvr = ipw + (size_t)(512 + tid) * DM;
    #pragma unroll 4
    for (int d2 = 0; d2 < DM; ++d2) cj += uu[h][d2] * wvr[d2];
    ctxs[tid] = cj;
    __syncthreads();

    // out-proj
    float oj = bo[tid];
    const float* wor = wo + (size_t)tid * DM;
    #pragma unroll 4
    for (int d2 = 0; d2 < DM; ++d2) oj += ctxs[d2] * wor[d2];

    // LayerNorm over 256
    float v1 = oj, v2 = oj * oj;
    #pragma unroll
    for (int off = 16; off > 0; off >>= 1) {
        v1 += __shfl_down(v1, off, 32);
        v2 += __shfl_down(v2, off, 32);
    }
    const int lane = tid & 31, w = tid >> 5;
    if (lane == 0) { red[w] = v1; red[8 + w] = v2; }
    __syncthreads();
    if (tid == 0) {
        float s1 = 0.f, s2 = 0.f;
        for (int i = 0; i < 8; ++i) { s1 += red[i]; s2 += red[8 + i]; }
        float mu = s1 * (1.f / 256.f);
        float var = s2 * (1.f / 256.f) - mu * mu;
        mu_s = mu;
        rs_s = rsqrtf(var + 1e-5f);
    }
    __syncthreads();
    out[obase + tid] = (oj - mu_s) * rs_s * gamma[tid] + beta[tid];
}

// ---------------------------------------------------------------------------
extern "C" void kernel_launch(void* const* d_in, const int* in_sizes, int n_in,
                              void* d_out, int out_size, void* d_ws, size_t ws_size,
                              hipStream_t stream) {
    const float*         x     = (const float*)d_in[0];
    const float*         query = (const float*)d_in[1];
    const float*         ipw   = (const float*)d_in[2];
    const float*         ipb   = (const float*)d_in[3];
    const float*         wo    = (const float*)d_in[4];
    const float*         bo    = (const float*)d_in[5];
    const float*         gamma = (const float*)d_in[6];
    const float*         beta  = (const float*)d_in[7];
    const unsigned char* mask  = (const unsigned char*)d_in[8];
    const int*           bid   = (const int*)d_in[9];
    float* ws  = (float*)d_ws;
    float* out = (float*)d_out;

    bqp_prep <<<1,            256, 0, stream>>>(query, ipw, ipb, ws);
    bqp_main <<<BX * NCHUNK,  256, 0, stream>>>(x, ipw, mask, bid, ws);
    bqp_final<<<NBK * BX,     256, 0, stream>>>(ipw, ipb, wo, bo, gamma, beta, ws, out);
}
